// MultiHeadAttentionBlock_55533927137599
// MI455X (gfx1250) — matrix operand order repaired and checked by reading, not compiled, not run on hardware
//
#include <hip/hip_runtime.h>
#include <hip/hip_bf16.h>

typedef __bf16 bf16_t;
typedef __attribute__((ext_vector_type(16))) __bf16 v16bf;
typedef __attribute__((ext_vector_type(8)))  __bf16 v8bf;
typedef __attribute__((ext_vector_type(8)))  float  v8f;

#define WMMA_BF16(a, b, c) \
  __builtin_amdgcn_wmma_f32_16x16x32_bf16(false, (a), false, (b), (short)0, (c), false, false)

// ---- gfx1250 async global->LDS copy (ASYNCcnt-tracked, bypasses VGPRs) ------
// GV mode: per-lane 64-bit global address, vdst VGPR holds LDS byte address.
static __device__ inline void async_cp16(unsigned lds_addr, const void* gaddr) {
  asm volatile("global_load_async_to_lds_b128 %0, %1, off"
               :: "v"(lds_addr), "v"((unsigned long long)(size_t)gaddr)
               : "memory");
}
static __device__ inline void wait_async0() {
  asm volatile("s_wait_asynccnt 0x0" ::: "memory");
}
static __device__ inline unsigned lds_off(const void* p) {
  // generic pointer to LDS: low 32 bits are the in-group LDS byte offset
  return (unsigned)(size_t)p;
}

// ---- WMMA fragment loaders from LDS -----------------------------------------
// A operand (16xK, 16-bit): lane m=L%16 holds row m; lanes 0-15 hold K[kb..kb+7]
// and K[kb+16..kb+23], lanes 16-31 hold K[kb+8..kb+15] and K[kb+24..kb+31].
static __device__ inline v16bf lds_load_a(const bf16_t* base, int stride,
                                          int rowbase, int kbase, int lane) {
  const int half = lane >> 4;
  const int m    = lane & 15;
  const bf16_t* p = base + (rowbase + m) * stride + kbase + half * 8;
  v8bf lo = *(const v8bf*)p;
  v8bf hi = *(const v8bf*)(p + 16);
  return __builtin_shufflevector(lo, hi, 0,1,2,3,4,5,6,7,8,9,10,11,12,13,14,15);
}

// B operand (Kx16, 16-bit): lane L holds column n=L%16; lanes 0-15 hold K[kb..kb+7]
// and K[kb+16..kb+23], lanes 16-31 hold K[kb+8..kb+15] and K[kb+24..kb+31].
static __device__ inline v16bf lds_load_b(const bf16_t* base, int stride,
                                          int nbase, int kbase, int lane) {
  const int n  = lane & 15;
  const int kk = kbase + (lane >> 4) * 8;
  const bf16_t* p = base + (nbase + n) * stride + kk;
  v8bf lo = *(const v8bf*)p;
  v8bf hi = *(const v8bf*)(p + 16);
  return __builtin_shufflevector(lo, hi, 0,1,2,3,4,5,6,7,8,9,10,11,12,13,14,15);
}

static __device__ inline v8bf cvt8(float4 a, float4 b) {
  v8bf r;
  r[0] = (__bf16)a.x; r[1] = (__bf16)a.y; r[2] = (__bf16)a.z; r[3] = (__bf16)a.w;
  r[4] = (__bf16)b.x; r[5] = (__bf16)b.y; r[6] = (__bf16)b.z; r[7] = (__bf16)b.w;
  return r;
}

// =============================================================================
// GEMM: C[M=8192, N=1024] = X[M,1024] @ W[N,1024]^T, bf16 WMMA, f32 accum.
// MODE 0: out bf16 [B,H,S,DK]        (Q, K projections, split heads)
// MODE 1: out bf16 [B,H,DK,S]        (V projection, head-transposed)
// MODE 2: out f32  [M,N] row-major   (final output projection)
// Block: 256 threads (8 waves), 128x128 tile, K-step 64 (16 WMMA/wave/phase).
// =============================================================================
template <int MODE, typename XT>
__global__ __launch_bounds__(256) void gemm128(const XT* __restrict__ X,
                                               const float* __restrict__ W,
                                               bf16_t* __restrict__ outb,
                                               float* __restrict__ outf) {
  constexpr int K = 1024;
  __shared__ __align__(64) bf16_t As[128 * 64];
  __shared__ __align__(64) bf16_t Bs[128 * 64];

  const int t     = threadIdx.x;
  const int lane  = t & 31;
  const int wid   = t >> 5;
  const int wm    = wid >> 2;          // 0..1 : 64-row strip
  const int wn    = wid & 3;           // 0..3 : 32-col strip
  const int mBase = blockIdx.x * 128;
  const int nBase = blockIdx.y * 128;
  const int lrow  = t >> 1;            // 0..127
  const int lcol  = (t & 1) * 32;      // 0 or 32

  v8f acc[4][2] = {};

  for (int k0 = 0; k0 < K; k0 += 64) {
    // stage X tile (convert f32 -> bf16 if needed)
    if constexpr (sizeof(XT) == 4) {
      const float4* xp = (const float4*)((const float*)X + (size_t)(mBase + lrow) * K + k0 + lcol);
      v8bf* d = (v8bf*)(As + lrow * 64 + lcol);
#pragma unroll
      for (int i = 0; i < 4; ++i) d[i] = cvt8(xp[2 * i], xp[2 * i + 1]);
    } else {
      const v8bf* xp = (const v8bf*)((const bf16_t*)X + (size_t)(mBase + lrow) * K + k0 + lcol);
      v8bf* d = (v8bf*)(As + lrow * 64 + lcol);
#pragma unroll
      for (int i = 0; i < 4; ++i) d[i] = xp[i];
    }
    // stage W tile (rows of W are contiguous in k: perfect B-operand layout)
    {
      const float4* wp = (const float4*)(W + (size_t)(nBase + lrow) * K + k0 + lcol);
      v8bf* d = (v8bf*)(Bs + lrow * 64 + lcol);
#pragma unroll
      for (int i = 0; i < 4; ++i) d[i] = cvt8(wp[2 * i], wp[2 * i + 1]);
    }
    __syncthreads();

#pragma unroll
    for (int kc = 0; kc < 2; ++kc) {
      v16bf a[4], b[2];
#pragma unroll
      for (int mt = 0; mt < 4; ++mt) a[mt] = lds_load_a(As, 64, wm * 64 + mt * 16, kc * 32, lane);
#pragma unroll
      for (int nt = 0; nt < 2; ++nt) b[nt] = lds_load_b(Bs, 64, wn * 32 + nt * 16, kc * 32, lane);
#pragma unroll
      for (int mt = 0; mt < 4; ++mt)
#pragma unroll
        for (int nt = 0; nt < 2; ++nt) acc[mt][nt] = WMMA_BF16(a[mt], b[nt], acc[mt][nt]);
    }
    __syncthreads();
  }

  // epilogue: C frag layout -> row = r + 8*(lane/16), col = lane%16
  const int half = lane >> 4, ln = lane & 15;
#pragma unroll
  for (int mt = 0; mt < 4; ++mt)
#pragma unroll
    for (int nt = 0; nt < 2; ++nt)
#pragma unroll
      for (int r = 0; r < 8; ++r) {
        const int m = mBase + wm * 64 + mt * 16 + r + 8 * half;
        const int n = nBase + wn * 32 + nt * 16 + ln;
        const float v = acc[mt][nt][r];
        if constexpr (MODE == 2) {
          outf[(size_t)m * 1024 + n] = v;
        } else {
          const int b_ = m >> 11, s_ = m & 2047;   // m = b*2048 + s
          const int h_ = n >> 6, dk = n & 63;      // n = h*64 + dk
          size_t idx;
          if constexpr (MODE == 0)
            idx = ((size_t)(b_ * 16 + h_) * 2048 + s_) * 64 + dk;
          else
            idx = ((size_t)(b_ * 16 + h_) * 64 + dk) * 2048 + s_;
          outb[idx] = (bf16_t)v;
        }
      }
}

// =============================================================================
// Causal flash attention. Grid: (S/128 = 16, B*H = 64). Block: 256 thr, 8 waves.
// K/V tiles double-buffered in LDS and fetched with global_load_async_to_lds
// (ASYNCcnt), prefetching tile j+1 while tile j is consumed by WMMAs.
// Qb/Kb: bf16 [bh][s][64]; Vt: bf16 [bh][64][s]; Ob: bf16 [b][s][h*64+dk].
// =============================================================================
__global__ __launch_bounds__(256) void attn_kernel(const bf16_t* __restrict__ Qb,
                                                   const bf16_t* __restrict__ Kb,
                                                   const bf16_t* __restrict__ Vt,
                                                   bf16_t* __restrict__ Ob) {
  __shared__ __align__(64) bf16_t Qs[128 * 64];
  __shared__ __align__(64) bf16_t Ks[2][128 * 64];
  __shared__ __align__(64) bf16_t Vs[2][64 * 128];    // V^T tile: [dk][key]
  __shared__ __align__(64) bf16_t Ps[8][16 * 144];    // per-wave P strip, padded

  const int t    = threadIdx.x;
  const int lane = t & 31;
  const int wid  = t >> 5;
  const int half = lane >> 4, ln = lane & 15;
  const int qt   = blockIdx.x;   // query tile
  const int bh   = blockIdx.y;   // head index
  const bf16_t* Kg = Kb + (size_t)bh * 2048 * 64;
  const bf16_t* Vg = Vt + (size_t)bh * 2048 * 64;

  // async-stage Q tile (64 B per thread = 4 x b128)
  {
    const int row = t >> 1, cb = (t & 1) * 32;
    const bf16_t* g = Qb + (size_t)bh * 2048 * 64 + (size_t)(qt * 128 + row) * 64 + cb;
    unsigned l = lds_off(Qs + row * 64 + cb);
#pragma unroll
    for (int i = 0; i < 4; ++i) async_cp16(l + 16 * i, (const char*)g + 16 * i);
  }

  // async-stage K/V tile j into buffer b
  auto issue_kv = [&](int j, int b) {
    const int row = t >> 1, cb = (t & 1) * 32;
    const bf16_t* kg = Kg + (size_t)(j * 128 + row) * 64 + cb;
    unsigned kl = lds_off(&Ks[b][row * 64 + cb]);
    const int dr = t >> 2, vb = (t & 3) * 32;
    const bf16_t* vg = Vg + (size_t)dr * 2048 + j * 128 + vb;
    unsigned vl = lds_off(&Vs[b][dr * 128 + vb]);
#pragma unroll
    for (int i = 0; i < 4; ++i) {
      async_cp16(kl + 16 * i, (const char*)kg + 16 * i);
      async_cp16(vl + 16 * i, (const char*)vg + 16 * i);
    }
  };

  issue_kv(0, 0);

  float m_st[8], l_st[8];
  v8f o[4] = {};
#pragma unroll
  for (int r = 0; r < 8; ++r) { m_st[r] = -1e30f; l_st[r] = 0.0f; }

  for (int j = 0; j <= qt; ++j) {
    const int buf = j & 1;
    wait_async0();        // own async loads (incl. Q on first pass) have landed
    __syncthreads();      // -> everyone's loads have landed; prev buf fully read
    if (j < qt) issue_kv(j + 1, buf ^ 1);   // prefetch overlaps compute below

    const bf16_t* KsB = &Ks[buf][0];
    const bf16_t* VsB = &Vs[buf][0];

    // S = Q K^T for this wave's 16-row strip (16 x 128), K-dim = 64 (2 chunks)
    v16bf aq0 = lds_load_a(Qs, 64, wid * 16, 0, lane);
    v16bf aq1 = lds_load_a(Qs, 64, wid * 16, 32, lane);
    v8f sfr[8];
#pragma unroll
    for (int ct = 0; ct < 8; ++ct) {
      v16bf b0 = lds_load_b(KsB, 64, ct * 16, 0, lane);
      v16bf b1 = lds_load_b(KsB, 64, ct * 16, 32, lane);
      v8f c = {};
      c = WMMA_BF16(aq0, b0, c);
      c = WMMA_BF16(aq1, b1, c);
      sfr[ct] = c;
    }

    // online softmax directly on C-fragment layout
    const int qrow0 = qt * 128 + wid * 16;
#pragma unroll
    for (int r = 0; r < 8; ++r) {
      const int qrow = qrow0 + r + 8 * half;
      float sv[8];
      float mloc = -1e30f;
#pragma unroll
      for (int ct = 0; ct < 8; ++ct) {
        const int key = j * 128 + ct * 16 + ln;
        float v = sfr[ct][r] * 0.125f;             // 1/sqrt(64)
        v = (key <= qrow) ? v : -1e30f;            // causal mask
        sv[ct] = v;
        mloc = fmaxf(mloc, v);
      }
#pragma unroll
      for (int off = 1; off < 16; off <<= 1)
        mloc = fmaxf(mloc, __shfl_xor(mloc, off, 32));
      const float mnew  = fmaxf(m_st[r], mloc);
      const float alpha = __expf(m_st[r] - mnew);
      float psum = 0.0f;
      bf16_t* prow = &Ps[wid][(r + 8 * half) * 144];
#pragma unroll
      for (int ct = 0; ct < 8; ++ct) {
        const float p = __expf(sv[ct] - mnew);
        psum += p;
        prow[ct * 16 + ln] = (bf16_t)p;            // re-layout via per-wave LDS
      }
#pragma unroll
      for (int off = 1; off < 16; off <<= 1) psum += __shfl_xor(psum, off, 32);
      l_st[r] = l_st[r] * alpha + psum;
      m_st[r] = mnew;
#pragma unroll
      for (int dt = 0; dt < 4; ++dt) o[dt][r] *= alpha;
    }

    // O += P @ V  (keys = 128 -> 4 chunks of 32; dk = 64 -> 4 col tiles)
#pragma unroll
    for (int c = 0; c < 4; ++c) {
      v16bf ap = lds_load_a(&Ps[wid][0], 144, 0, c * 32, lane);
#pragma unroll
      for (int dt = 0; dt < 4; ++dt) {
        v16bf bv = lds_load_b(VsB, 128, dt * 16, c * 32, lane);
        o[dt] = WMMA_BF16(ap, bv, o[dt]);
      }
    }
    // no trailing barrier: next iteration's wait+barrier protects buffer reuse
  }

  // normalize and write merged-head bf16 output [b][s][h*64+dk]
  const int b_ = bh >> 4, h_ = bh & 15;
#pragma unroll
  for (int dt = 0; dt < 4; ++dt)
#pragma unroll
    for (int r = 0; r < 8; ++r) {
      const float val = o[dt][r] / l_st[r];
      const int s_ = qt * 128 + wid * 16 + r + 8 * half;
      const int dk = dt * 16 + ln;
      Ob[((size_t)(b_ * 2048 + s_)) * 1024 + h_ * 64 + dk] = (bf16_t)val;
    }
}

// =============================================================================
extern "C" void kernel_launch(void* const* d_in, const int* in_sizes, int n_in,
                              void* d_out, int out_size, void* d_ws, size_t ws_size,
                              hipStream_t stream) {
  const float* q  = (const float*)d_in[0];
  const float* k  = (const float*)d_in[1];
  const float* v  = (const float*)d_in[2];
  // d_in[3] = mask (causal, implemented analytically)
  const float* Wq = (const float*)d_in[4];
  const float* Wk = (const float*)d_in[5];
  const float* Wv = (const float*)d_in[6];
  const float* Wo = (const float*)d_in[7];
  float* out = (float*)d_out;

  const size_t ELT = (size_t)4 * 2048 * 1024;   // elements per [B,S,D] tensor
  bf16_t* Qb = (bf16_t*)d_ws;                   // [bh][s][dk]
  bf16_t* Kb = Qb + ELT;                        // [bh][s][dk]
  bf16_t* Vt = Kb + ELT;                        // [bh][dk][s]
  bf16_t* Ob = Vt + ELT;                        // [b][s][d]

  dim3 gg(64, 8), gb(256);
  gemm128<0, float><<<gg, gb, 0, stream>>>(q, Wq, Qb, nullptr);
  gemm128<0, float><<<gg, gb, 0, stream>>>(k, Wk, Kb, nullptr);
  gemm128<1, float><<<gg, gb, 0, stream>>>(v, Wv, Vt, nullptr);
  attn_kernel<<<dim3(16, 64), gb, 0, stream>>>(Qb, Kb, Vt, Ob);
  gemm128<2, bf16_t><<<gg, gb, 0, stream>>>(Ob, Wo, nullptr, out);
}
